// SemanticsModulatedAttention_62878321214090
// MI455X (gfx1250) — matrix operand order repaired and checked by reading, not compiled, not run on hardware
//
#include <hip/hip_runtime.h>
#include <hip/hip_bf16.h>
#include <stdint.h>

#define B_  32
#define T_  1024
#define D_  512
#define DT_ 256
#define H_  8
#define NT_ 77
#define N_  2125
#define TE_ 2048
#define NEG_ (-1000000.0f)

typedef __bf16 bf16_t;
typedef __attribute__((ext_vector_type(16))) __bf16 v16bf;
typedef __attribute__((ext_vector_type(8)))  float  v8f;
typedef __attribute__((ext_vector_type(4)))  unsigned uint4v;
typedef __attribute__((ext_vector_type(8)))  int      int8v;
typedef __attribute__((ext_vector_type(4)))  int      int4v;

__device__ __forceinline__ bf16_t f2bf(float f) {
    unsigned u = __builtin_bit_cast(unsigned, f);
    unsigned r = u + 0x7FFFu + ((u >> 16) & 1u);
    return __builtin_bit_cast(bf16_t, (unsigned short)(r >> 16));
}
__device__ __forceinline__ float bf2f(bf16_t b) {
    unsigned u = ((unsigned)__builtin_bit_cast(unsigned short, b)) << 16;
    return __builtin_bit_cast(float, u);
}
__device__ __forceinline__ float sigm(float x) { return 1.0f / (1.0f + __expf(-x)); }

__device__ __forceinline__ float blockSum128(float v) {
    __shared__ float s[128];
    int t = threadIdx.x;
    s[t] = v; __syncthreads();
    for (int o = 64; o > 0; o >>= 1) { if (t < o) s[t] += s[t + o]; __syncthreads(); }
    float r = s[0]; __syncthreads();
    return r;
}

// ---------------- small prep kernels ----------------

__global__ void cond_kernel(const int* __restrict__ ct, float* __restrict__ text_ct,
                            float* __restrict__ retr_ct) {
    int b = threadIdx.x;
    if (b < B_) {
        int v = ct[b];
        text_ct[b] = ((v % 10) > 0) ? 1.0f : 0.0f;
        retr_ct[b] = ((v / 10) > 0) ? 1.0f : 0.0f;
    }
}

__global__ void cvt_bf16_kernel(const float* __restrict__ s, bf16_t* __restrict__ d, int n) {
    int i = blockIdx.x * 256 + threadIdx.x;
    if (i < n) d[i] = f2bf(s[i]);
}

// Row-wise LayerNorm -> bf16, C in {256,512}, block = 128
__global__ __launch_bounds__(128) void ln_rows_kernel(const float* __restrict__ x,
                                                      const float* __restrict__ g,
                                                      const float* __restrict__ bb,
                                                      bf16_t* __restrict__ out, int C) {
    int row = blockIdx.x;
    const float* src = x + (size_t)row * C;
    int per = C >> 7;
    float v[4];
    float s = 0.f;
    for (int i = 0; i < per; ++i) { v[i] = src[threadIdx.x + i * 128]; s += v[i]; }
    float mean = blockSum128(s) / (float)C;
    float q = 0.f;
    for (int i = 0; i < per; ++i) { float d = v[i] - mean; q += d * d; }
    float var = blockSum128(q) / (float)C;
    float inv = rsqrtf(var + 1e-5f);
    for (int i = 0; i < per; ++i) {
        int c = threadIdx.x + i * 128;
        out[(size_t)row * C + c] = f2bf((v[i] - mean) * inv * g[c] + bb[c]);
    }
}

// LN over concat(re_motion[row], re_text[b,rr]) (C = 1024) -> bf16
__global__ __launch_bounds__(128) void rfk_ln_kernel(const float* __restrict__ re_motion,
                                                     const float* __restrict__ re_text,
                                                     const float* __restrict__ g,
                                                     const float* __restrict__ bb,
                                                     bf16_t* __restrict__ out) {
    int row = blockIdx.x;           // 0 .. B*1024-1
    int b = row >> 10, m = row & 1023, rr = m >> 8;
    const float* pm = re_motion + (size_t)row * D_;
    const float* pt = re_text + (size_t)(b * 4 + rr) * D_;
    float v[8]; float s = 0.f;
    for (int i = 0; i < 8; ++i) {
        int c = threadIdx.x + i * 128;
        v[i] = (c < 512) ? pm[c] : pt[c - 512];
        s += v[i];
    }
    float mean = blockSum128(s) * (1.0f / 1024.0f);
    float q = 0.f;
    for (int i = 0; i < 8; ++i) { float d = v[i] - mean; q += d * d; }
    float var = blockSum128(q) * (1.0f / 1024.0f);
    float inv = rsqrtf(var + 1e-5f);
    for (int i = 0; i < 8; ++i) {
        int c = threadIdx.x + i * 128;
        out[(size_t)row * 1024 + c] = f2bf((v[i] - mean) * inv * g[c] + bb[c]);
    }
}

__global__ void siluemb_kernel(const float* __restrict__ e, float* __restrict__ o, int n) {
    int i = blockIdx.x * 256 + threadIdx.x;
    if (i < n) { float x = e[i]; o[i] = x * sigm(x); }
}

// emb_out[b, c] = silu(emb[b]) . emb_W[c] + emb_b[c]   (32 x 1024, K=2048; tiny fp32 GEMM)
__global__ __launch_bounds__(128) void embgemm_kernel(const float* __restrict__ se,
                                                      const float* __restrict__ W,
                                                      const float* __restrict__ bias,
                                                      float* __restrict__ out) {
    int b = blockIdx.x;
    int c = blockIdx.y * 128 + threadIdx.x;
    const float* e = se + (size_t)b * TE_;
    const float* w = W + (size_t)c * TE_;
    float s = 0.f;
    for (int k = 0; k < TE_; ++k) s += e[k] * w[k];
    out[(size_t)b * 1024 + c] = s + bias[c];
}

// --- TDM helper: stage W panel chunk [128 rows x 64 K] (bf16) into LDS via Tensor Data Mover
__device__ __forceinline__ void tdm_stage_w(const bf16_t* W, int K, int rowPanel, int k0,
                                            unsigned ldsByteOff) {
    unsigned long long ga = (unsigned long long)(const void*)W
                          + ((unsigned long long)rowPanel * (unsigned)K + (unsigned)k0) * 2ull;
    uint4v g0;
    g0.x = 1u;                                           // count=1, user descriptor
    g0.y = ldsByteOff;                                   // lds_addr (bytes)
    g0.z = (unsigned)(ga & 0xffffffffu);                 // global_addr[31:0]
    g0.w = (unsigned)((ga >> 32) & 0x01ffffffu) | (2u << 30);  // global_addr[56:32], type=2
    int8v g1;
    g1[0] = 0x10000;                                     // data_size=1 (2 bytes)
    g1[1] = (int)(((unsigned)K & 0xffffu) << 16);        // tensor_dim0 lo16 (= K)
    g1[2] = (int)((512u & 0xffffu) << 16) | (int)(((unsigned)K >> 16) & 0xffffu); // dim0 hi16 | dim1 lo16
    g1[3] = (int)(64u << 16);                            // tensor_dim1 hi16=0 | tile_dim0=64
    g1[4] = 128;                                         // tile_dim1=128, tile_dim2=0
    g1[5] = K;                                           // tensor_dim0_stride lo32
    g1[6] = 0;                                           // stride hi16 | dim1_stride lo16
    g1[7] = 0;                                           // dim1_stride hi32
    int4v gz = {0, 0, 0, 0};
#if defined(__clang_major__) && (__clang_major__ >= 23)
    int8v gz8 = {0, 0, 0, 0, 0, 0, 0, 0};
    __builtin_amdgcn_tensor_load_to_lds(g0, g1, gz, gz, gz8, 0);
#else
    __builtin_amdgcn_tensor_load_to_lds(g0, g1, gz, gz, 0);
#endif
}

// ---------------- generic WMMA GEMM: C = A(bf16)[Mb,K] x W(bf16)[512,K]^T + bias, epilogue ----
// W panel ([blockIdx.x*128 .. +128) rows x K) is double-buffer streamed into LDS by the TDM.
// modes: 0 none | 1 +(1-ct)*NEG | 2 +(1-ct)*NEG+(1-rowmask)*NEG | 3 +(1-rowmask)*NEG
//        4 *ct | 5 *ct*rowmask | 6 *rowmask | 7 +resid
__global__ __launch_bounds__(128) void wmma_gemm_kernel(
    const bf16_t* __restrict__ A, long aBatch, int Mb, int K,
    const bf16_t* __restrict__ W, const float* __restrict__ bias,
    float* outF, bf16_t* outB, long outBatch, int rowOff,
    int mode, const float* __restrict__ ct, const float* __restrict__ rowmask, int rms,
    const float* __restrict__ resid)
{
    extern __shared__ bf16_t ldsW[];   // 2 buffers x 128*64 bf16 = 32 KB, offset 0 in LDS

    const int b    = blockIdx.z;
    const int wid  = threadIdx.x >> 5;
    const int lane = threadIdx.x & 31;
    const int lr   = lane & 15, half = lane >> 4;
    const int mBase = blockIdx.y * 128 + (wid >> 1) * 64;
    const int nBase = blockIdx.x * 128 + (wid & 1) * 64;
    const int rowPanel = blockIdx.x * 128;
    const int nLocal = (wid & 1) * 64;
    const bf16_t* Ab = A + (size_t)b * aBatch;

    const v8f vzero = {0.f,0.f,0.f,0.f,0.f,0.f,0.f,0.f};
    v8f acc[4][4];
#pragma unroll
    for (int i = 0; i < 4; ++i)
#pragma unroll
        for (int j = 0; j < 4; ++j) acc[i][j] = vzero;

    // prologue: DMA chunk 0 into buffer 0
    if (wid == 0) {
        tdm_stage_w(W, K, rowPanel, 0, 0u);
        __builtin_amdgcn_s_wait_tensorcnt(0);
    }
    __syncthreads();

    int cur = 0;
    for (int k0 = 0; k0 < K; k0 += 64) {
        // kick off DMA for next chunk into the other buffer (overlaps with compute)
        if (wid == 0 && (k0 + 64) < K)
            tdm_stage_w(W, K, rowPanel, k0 + 64, (unsigned)((cur ^ 1) * 128 * 64 * 2));

        const bf16_t* Wp = ldsW + (size_t)cur * (128 * 64);
#pragma unroll
        for (int kk = 0; kk < 64; kk += 32) {
            v16bf fa[4], fb[4];
#pragma unroll
            for (int i = 0; i < 4; ++i) {
                int row = mBase + i * 16 + lr; if (row >= Mb) row = Mb - 1;
                const bf16_t* p = Ab + (size_t)row * K + k0 + kk + half * 8;
                ((uint4*)&fa[i])[0] = *(const uint4*)p;
                ((uint4*)&fa[i])[1] = *(const uint4*)(p + 16);
                if (kk == 0 && (k0 + 64) < K)
                    __builtin_prefetch((const void*)(p + 64), 0, 3);
            }
#pragma unroll
            for (int j = 0; j < 4; ++j) {
                const bf16_t* p = Wp + (size_t)(nLocal + j * 16 + lr) * 64 + kk + half * 8;
                ((uint4*)&fb[j])[0] = *(const uint4*)p;
                ((uint4*)&fb[j])[1] = *(const uint4*)(p + 16);
            }
#pragma unroll
            for (int i = 0; i < 4; ++i)
#pragma unroll
                for (int j = 0; j < 4; ++j)
                    acc[i][j] = __builtin_amdgcn_wmma_f32_16x16x32_bf16(
                        false, fa[i], false, fb[j], (short)0, acc[i][j], false, false);
        }
        // next buffer must be complete before anyone reads it
        if (wid == 0 && (k0 + 64) < K) __builtin_amdgcn_s_wait_tensorcnt(0);
        __syncthreads();
        cur ^= 1;
    }

#pragma unroll
    for (int j = 0; j < 4; ++j) {
        const int c = nBase + j * 16 + lr;
        const float bc = bias[c];
#pragma unroll
        for (int i = 0; i < 4; ++i) {
            const float* ap = (const float*)&acc[i][j];
            const int rb = mBase + i * 16 + 8 * half;
#pragma unroll
            for (int v = 0; v < 8; ++v) {
                int r = rb + v;
                if (r >= Mb) continue;
                float val = ap[v] + bc;
                if      (mode == 1) val += (1.f - ct[b]) * NEG_;
                else if (mode == 2) val += (1.f - ct[b]) * NEG_ + (1.f - rowmask[(size_t)b * rms + r]) * NEG_;
                else if (mode == 3) val += (1.f - rowmask[(size_t)b * rms + r]) * NEG_;
                else if (mode == 4) val *= ct[b];
                else if (mode == 5) val *= ct[b] * rowmask[(size_t)b * rms + r];
                else if (mode == 6) val *= rowmask[(size_t)b * rms + r];
                size_t oi = (size_t)b * outBatch + (size_t)(rowOff + r) * D_ + c;
                if (mode == 7) val += resid[oi];
                if (outF) outF[oi] = val; else outB[oi] = f2bf(val);
            }
        }
    }
}

// ---------------- softmaxes ----------------

// softmax over N per (b, channel d); key is fp32 [B,N,512]; writes bf16
__global__ void ksoftmax_kernel(float* __restrict__ key, bf16_t* __restrict__ kbf) {
    int b = blockIdx.x;
    int d = blockIdx.y * 256 + threadIdx.x;
    float* col = key + (size_t)b * N_ * D_ + d;
    float mx = -3.4e38f;
    for (int n = 0; n < N_; ++n) mx = fmaxf(mx, col[(size_t)n * D_]);
    float s = 0.f;
    for (int n = 0; n < N_; ++n) { float e = __expf(col[(size_t)n * D_] - mx); col[(size_t)n * D_] = e; s += e; }
    float inv = 1.f / s;
    bf16_t* o = kbf + (size_t)b * N_ * D_ + d;
    for (int n = 0; n < N_; ++n) o[(size_t)n * D_] = f2bf(col[(size_t)n * D_] * inv);
}

// softmax over head_dim (64) per (b,t,h); one wave per row
__global__ void qsoftmax_kernel(const float* __restrict__ q, bf16_t* __restrict__ qbf) {
    int wid = threadIdx.x >> 5, lane = threadIdx.x & 31;
    size_t row = (size_t)blockIdx.x * 8 + wid;
    const float* src = q + row * 64;
    float a = src[lane], c = src[lane + 32];
    float mx = fmaxf(a, c);
    for (int o = 16; o > 0; o >>= 1) mx = fmaxf(mx, __shfl_xor(mx, o, 32));
    float ea = __expf(a - mx), ec = __expf(c - mx);
    float s = ea + ec;
    for (int o = 16; o > 0; o >>= 1) s += __shfl_xor(s, o, 32);
    float inv = 1.f / s;
    qbf[row * 64 + lane]      = f2bf(ea * inv);
    qbf[row * 64 + lane + 32] = f2bf(ec * inv);
}

// ---------------- attention = K^T V per (b,h), 64x64, WMMA over N chunks ----------------
__global__ __launch_bounds__(128) void attn_kernel(const bf16_t* __restrict__ kbf,
                                                   const bf16_t* __restrict__ vbf,
                                                   bf16_t* __restrict__ attT) {
    __shared__ __align__(16) bf16_t KT[4][64 * 32];
    __shared__ __align__(16) bf16_t VT[4][64 * 32];
    __shared__ float accS[64 * 64];

    const int bh = blockIdx.x;
    const int b = bh >> 3, h = bh & 7;
    const int hoff = h * 64;
    const int wid = threadIdx.x >> 5;
    const int lane = threadIdx.x & 31;
    const int lr = lane & 15, half = lane >> 4;

    for (int e = threadIdx.x; e < 4096; e += 128) accS[e] = 0.f;
    __syncthreads();

    const v8f vzero = {0.f,0.f,0.f,0.f,0.f,0.f,0.f,0.f};
    v8f acc[4][4];
#pragma unroll
    for (int i = 0; i < 4; ++i)
#pragma unroll
        for (int j = 0; j < 4; ++j) acc[i][j] = vzero;

    const int nchunks = (N_ + 31) / 32;      // 67
    const int iters = (nchunks + 3) / 4;     // 17 (uniform over waves)
    for (int it = 0; it < iters; ++it) {
        int c = it * 4 + wid;
        int nb = c * 32;
        for (int n = 0; n < 32; ++n) {       // stage transposed [d][n] tiles
            int gn = nb + n;
            unsigned kk = 0, vv = 0;
            if (c < nchunks && gn < N_) {
                size_t off = ((size_t)b * N_ + gn) * D_ + hoff + 2 * lane;
                kk = *(const unsigned*)(kbf + off);
                vv = *(const unsigned*)(vbf + off);
            }
            KT[wid][(2 * lane) * 32 + n]     = __builtin_bit_cast(bf16_t, (unsigned short)(kk & 0xffff));
            KT[wid][(2 * lane + 1) * 32 + n] = __builtin_bit_cast(bf16_t, (unsigned short)(kk >> 16));
            VT[wid][(2 * lane) * 32 + n]     = __builtin_bit_cast(bf16_t, (unsigned short)(vv & 0xffff));
            VT[wid][(2 * lane + 1) * 32 + n] = __builtin_bit_cast(bf16_t, (unsigned short)(vv >> 16));
        }
        __syncthreads();
        v16bf fa[4], fb[4];
#pragma unroll
        for (int i = 0; i < 4; ++i) {
            const bf16_t* p = &KT[wid][(i * 16 + lr) * 32 + half * 8];
            ((uint4*)&fa[i])[0] = *(const uint4*)p;
            ((uint4*)&fa[i])[1] = *(const uint4*)(p + 16);
        }
#pragma unroll
        for (int j = 0; j < 4; ++j) {
            const bf16_t* p = &VT[wid][(j * 16 + lr) * 32 + half * 8];
            ((uint4*)&fb[j])[0] = *(const uint4*)p;
            ((uint4*)&fb[j])[1] = *(const uint4*)(p + 16);
        }
#pragma unroll
        for (int i = 0; i < 4; ++i)
#pragma unroll
            for (int j = 0; j < 4; ++j)
                acc[i][j] = __builtin_amdgcn_wmma_f32_16x16x32_bf16(
                    false, fa[i], false, fb[j], (short)0, acc[i][j], false, false);
        __syncthreads();
    }

#pragma unroll
    for (int i = 0; i < 4; ++i)
#pragma unroll
        for (int j = 0; j < 4; ++j) {
            const float* ap = (const float*)&acc[i][j];
#pragma unroll
            for (int v = 0; v < 8; ++v)
                atomicAdd(&accS[(i * 16 + 8 * half + v) * 64 + (j * 16 + lr)], ap[v]);
        }
    __syncthreads();
    for (int e = threadIdx.x; e < 4096; e += 128) {
        int l = e >> 6, d = e & 63;
        attT[((size_t)bh * 64 + l) * 64 + d] = f2bf(accS[d * 64 + l]);  // stored transposed
    }
}

// ---------------- y = Q (softmaxed, bf16) x att per (b,h) ----------------
__global__ __launch_bounds__(128) void ygemm_kernel(const bf16_t* __restrict__ qbf,
                                                    const bf16_t* __restrict__ attT,
                                                    float* __restrict__ y) {
    const int bh = blockIdx.x;
    const int b = bh >> 3, h = bh & 7;
    const int hoff = h * 64;
    const int wid = threadIdx.x >> 5, lane = threadIdx.x & 31;
    const int lr = lane & 15, half = lane >> 4;
    const int tBase = blockIdx.y * 256 + wid * 64;

    const v8f vzero = {0.f,0.f,0.f,0.f,0.f,0.f,0.f,0.f};
    v8f acc[4][4];
#pragma unroll
    for (int i = 0; i < 4; ++i)
#pragma unroll
        for (int j = 0; j < 4; ++j) acc[i][j] = vzero;

#pragma unroll
    for (int k0 = 0; k0 < 64; k0 += 32) {
        v16bf fa[4], fb[4];
#pragma unroll
        for (int i = 0; i < 4; ++i) {
            int t = tBase + i * 16 + lr;
            const bf16_t* p = qbf + ((size_t)(b * T_ + t) * D_ + hoff + k0 + half * 8);
            ((uint4*)&fa[i])[0] = *(const uint4*)p;
            ((uint4*)&fa[i])[1] = *(const uint4*)(p + 16);
        }
#pragma unroll
        for (int j = 0; j < 4; ++j) {
            const bf16_t* p = attT + ((size_t)bh * 64 + (j * 16 + lr)) * 64 + k0 + half * 8;
            ((uint4*)&fb[j])[0] = *(const uint4*)p;
            ((uint4*)&fb[j])[1] = *(const uint4*)(p + 16);
        }
#pragma unroll
        for (int i = 0; i < 4; ++i)
#pragma unroll
            for (int j = 0; j < 4; ++j)
                acc[i][j] = __builtin_amdgcn_wmma_f32_16x16x32_bf16(
                    false, fa[i], false, fb[j], (short)0, acc[i][j], false, false);
    }

#pragma unroll
    for (int j = 0; j < 4; ++j) {
        int c = hoff + j * 16 + lr;
#pragma unroll
        for (int i = 0; i < 4; ++i) {
            const float* ap = (const float*)&acc[i][j];
            int rb = tBase + i * 16 + 8 * half;
#pragma unroll
            for (int v = 0; v < 8; ++v)
                y[(size_t)(b * T_ + rb + v) * D_ + c] = ap[v];
        }
    }
}

// ---------------- modulated LN + silu -> bf16 ----------------
__global__ __launch_bounds__(128) void moduln_kernel(const float* __restrict__ y,
                                                     const float* __restrict__ embout,
                                                     const float* __restrict__ g,
                                                     const float* __restrict__ bb,
                                                     bf16_t* __restrict__ out) {
    int row = blockIdx.x;
    int b = row >> 10;
    const float* src = y + (size_t)row * D_;
    float v[4]; float s = 0.f;
    for (int i = 0; i < 4; ++i) { v[i] = src[threadIdx.x + i * 128]; s += v[i]; }
    float mean = blockSum128(s) * (1.0f / D_);
    float q = 0.f;
    for (int i = 0; i < 4; ++i) { float d = v[i] - mean; q += d * d; }
    float var = blockSum128(q) * (1.0f / D_);
    float inv = rsqrtf(var + 1e-5f);
    for (int i = 0; i < 4; ++i) {
        int c = threadIdx.x + i * 128;
        float n = (v[i] - mean) * inv * g[c] + bb[c];
        float sc = embout[(size_t)b * 1024 + c];
        float sh = embout[(size_t)b * 1024 + 512 + c];
        float hh = n * (1.f + sc) + sh;
        out[(size_t)row * D_ + c] = f2bf(hh * sigm(hh));
    }
}

// ---------------- host launch ----------------
extern "C" void kernel_launch(void* const* d_in, const int* in_sizes, int n_in,
                              void* d_out, int out_size, void* d_ws, size_t ws_size,
                              hipStream_t stream) {
    const float* x         = (const float*)d_in[0];
    const float* xf        = (const float*)d_in[1];
    const float* emb       = (const float*)d_in[2];
    const float* src_mask  = (const float*)d_in[3];
    const int*   cond_type = (const int*)d_in[4];
    const float* re_motion = (const float*)d_in[5];
    const float* re_text   = (const float*)d_in[6];
    const float* re_mask   = (const float*)d_in[7];
    const float* norm_g = (const float*)d_in[8],  *norm_b = (const float*)d_in[9];
    const float* tnorm_g = (const float*)d_in[10], *tnorm_b = (const float*)d_in[11];
    const float* rn1_g = (const float*)d_in[12], *rn1_b = (const float*)d_in[13];
    const float* rn2_g = (const float*)d_in[14], *rn2_b = (const float*)d_in[15];
    const float* Wq  = (const float*)d_in[16], *bq  = (const float*)d_in[17];
    const float* Wkt = (const float*)d_in[18], *bkt = (const float*)d_in[19];
    const float* Wvt = (const float*)d_in[20], *bvt = (const float*)d_in[21];
    const float* Wkm = (const float*)d_in[22], *bkm = (const float*)d_in[23];
    const float* Wvm = (const float*)d_in[24], *bvm = (const float*)d_in[25];
    const float* Wkr = (const float*)d_in[26], *bkr = (const float*)d_in[27];
    const float* Wvr = (const float*)d_in[28], *bvr = (const float*)d_in[29];
    const float* emb_W = (const float*)d_in[30], *emb_b = (const float*)d_in[31];
    const float* snorm_g = (const float*)d_in[32], *snorm_b = (const float*)d_in[33];
    const float* out_W = (const float*)d_in[34], *out_b = (const float*)d_in[35];
    float* out = (float*)d_out;

    char* p = (char*)d_ws;
    auto take = [&](size_t bytes) -> char* {
        char* r = p; p += (bytes + 255) & ~(size_t)255; return r;
    };
    bf16_t* xn_bf  = (bf16_t*)take((size_t)B_ * T_ * D_ * 2);
    bf16_t* xfn_bf = (bf16_t*)take((size_t)B_ * NT_ * DT_ * 2);
    bf16_t* rfk_bf = (bf16_t*)take((size_t)B_ * 1024 * 1024 * 2);
    bf16_t* rfv_bf = (bf16_t*)take((size_t)B_ * 1024 * 512 * 2);
    bf16_t* wq_bf  = (bf16_t*)take(512 * 512 * 2);
    bf16_t* wkt_bf = (bf16_t*)take(512 * 256 * 2);
    bf16_t* wvt_bf = (bf16_t*)take(512 * 256 * 2);
    bf16_t* wkm_bf = (bf16_t*)take(512 * 512 * 2);
    bf16_t* wvm_bf = (bf16_t*)take(512 * 512 * 2);
    bf16_t* wkr_bf = (bf16_t*)take(512 * 1024 * 2);
    bf16_t* wvr_bf = (bf16_t*)take(512 * 512 * 2);
    bf16_t* wo_bf  = (bf16_t*)take(512 * 512 * 2);
    float*  queryY = (float*)take((size_t)B_ * T_ * D_ * 4);   // query, later reused as y
    bf16_t* q_bf   = (bf16_t*)take((size_t)B_ * T_ * D_ * 2);
    float*  keyf   = (float*)take((size_t)B_ * N_ * D_ * 4);
    bf16_t* k_bf   = (bf16_t*)take((size_t)B_ * N_ * D_ * 2);
    bf16_t* v_bf   = (bf16_t*)take((size_t)B_ * N_ * D_ * 2);
    bf16_t* attT   = (bf16_t*)take((size_t)B_ * H_ * 64 * 64 * 2);
    float*  embout = (float*)take((size_t)B_ * 1024 * 4);
    float*  silue  = (float*)take((size_t)B_ * TE_ * 4);
    float*  text_ct = (float*)take(B_ * 4);
    float*  retr_ct = (float*)take(B_ * 4);
    bf16_t* siluh = xn_bf;   // reuse: xn consumed by v_mot GEMM before moduln writes this

    cond_kernel<<<1, 32, 0, stream>>>(cond_type, text_ct, retr_ct);

    auto cvt = [&](const float* s, bf16_t* d, int n) {
        cvt_bf16_kernel<<<(n + 255) / 256, 256, 0, stream>>>(s, d, n);
    };
    cvt(Wq,  wq_bf,  512 * 512);
    cvt(Wkt, wkt_bf, 512 * 256);
    cvt(Wvt, wvt_bf, 512 * 256);
    cvt(Wkm, wkm_bf, 512 * 512);
    cvt(Wvm, wvm_bf, 512 * 512);
    cvt(Wkr, wkr_bf, 512 * 1024);
    cvt(Wvr, wvr_bf, 512 * 512);
    cvt(out_W, wo_bf, 512 * 512);

    ln_rows_kernel<<<B_ * T_, 128, 0, stream>>>(x, norm_g, norm_b, xn_bf, 512);
    ln_rows_kernel<<<B_ * NT_, 128, 0, stream>>>(xf, tnorm_g, tnorm_b, xfn_bf, 256);
    rfk_ln_kernel<<<B_ * 1024, 128, 0, stream>>>(re_motion, re_text, rn1_g, rn1_b, rfk_bf);
    ln_rows_kernel<<<B_ * 1024, 128, 0, stream>>>(re_motion, rn2_g, rn2_b, rfv_bf, 512);

    siluemb_kernel<<<(B_ * TE_ + 255) / 256, 256, 0, stream>>>(emb, silue, B_ * TE_);
    embgemm_kernel<<<dim3(B_, 8), 128, 0, stream>>>(silue, emb_W, emb_b, embout);

    const size_t ldsBytes = 2 * 128 * 64 * sizeof(bf16_t);   // 32 KB double buffer
    auto gemm = [&](const bf16_t* A, long aBatch, int Mb, int K, const bf16_t* Wb,
                    const float* bias, float* outF, bf16_t* outB, long outBatch, int rowOff,
                    int mode, const float* ctp, const float* rowmask, int rms,
                    const float* resid) {
        dim3 grid(D_ / 128, (Mb + 127) / 128, B_);
        wmma_gemm_kernel<<<grid, 128, ldsBytes, stream>>>(A, aBatch, Mb, K, Wb, bias, outF, outB,
                                                          outBatch, rowOff, mode, ctp, rowmask, rms, resid);
    };

    // query + key/value projections (epilogue applies cond/NEG masks, value written bf16)
    gemm(xn_bf,  (long)T_ * D_,   T_,   512,  wq_bf,  bq,  queryY, nullptr, (long)T_ * D_, 0,    0, nullptr, nullptr, 0, nullptr);
    gemm(xfn_bf, (long)NT_ * DT_, NT_,  256,  wkt_bf, bkt, keyf,   nullptr, (long)N_ * D_, 0,    1, text_ct, nullptr, 0, nullptr);
    gemm(rfk_bf, 1024L * 1024,    1024, 1024, wkr_bf, bkr, keyf,   nullptr, (long)N_ * D_, 77,   2, retr_ct, re_mask, 1024, nullptr);
    gemm(xn_bf,  (long)T_ * D_,   1024, 512,  wkm_bf, bkm, keyf,   nullptr, (long)N_ * D_, 1101, 3, nullptr, src_mask, 1024, nullptr);
    gemm(xfn_bf, (long)NT_ * DT_, NT_,  256,  wvt_bf, bvt, nullptr, v_bf,   (long)N_ * D_, 0,    4, text_ct, nullptr, 0, nullptr);
    gemm(rfv_bf, 1024L * 512,     1024, 512,  wvr_bf, bvr, nullptr, v_bf,   (long)N_ * D_, 77,   5, retr_ct, re_mask, 1024, nullptr);
    gemm(xn_bf,  (long)T_ * D_,   1024, 512,  wvm_bf, bvm, nullptr, v_bf,   (long)N_ * D_, 1101, 6, nullptr, src_mask, 1024, nullptr);

    ksoftmax_kernel<<<dim3(B_, 2), 256, 0, stream>>>(keyf, k_bf);
    qsoftmax_kernel<<<B_ * T_ * H_ / 8, 256, 0, stream>>>(queryY, q_bf);

    attn_kernel<<<B_ * H_, 128, 0, stream>>>(k_bf, v_bf, attT);
    ygemm_kernel<<<dim3(B_ * H_, T_ / 256), 128, 0, stream>>>(q_bf, attT, queryY); // queryY now = y

    moduln_kernel<<<B_ * T_, 128, 0, stream>>>(queryY, embout, snorm_g, snorm_b, siluh);

    // out = x + silu(h) @ out_W^T + out_b
    gemm(siluh, (long)T_ * D_, 1024, 512, wo_bf, out_b, out, nullptr, (long)T_ * D_, 0,
         7, nullptr, nullptr, 0, x);
}
